// GraphEncoder_32521492365955
// MI455X (gfx1250) — compile-verified, matrix operand order checked by reading, not run on hardware
//
#include <hip/hip_runtime.h>
#include <hip/hip_bf16.h>
#include <hip/hip_fp16.h>

// GAT graph encoder for MI455X (gfx1250, wave32).
// Dense projections: V_WMMA_F32_16X16X32_F16 with pre-packed f16 fragments
// (two global_load_b128 per operand per WMMA), A fragments held in VGPRs
// across all N-tiles, constexpr-K fully unrolled.
// Edge softmax/aggregation: ordered-uint GLOBAL_ATOMIC_MAX_U32 +
// GLOBAL_ATOMIC_ADD_F32, three passes.

typedef __attribute__((ext_vector_type(16))) _Float16 v16h;
typedef __attribute__((ext_vector_type(8)))  float    v8f;
typedef __attribute__((ext_vector_type(8)))  unsigned v8u;
typedef __attribute__((ext_vector_type(4)))  unsigned v4u;

#define SLOPE 0.2f
#define CEILDIV(a,b) (((a)+(b)-1)/(b))

__device__ __forceinline__ unsigned f2ord(float f) {
    unsigned u = __float_as_uint(f);
    return (u & 0x80000000u) ? ~u : (u | 0x80000000u);
}
__device__ __forceinline__ float ord2f(unsigned u) {
    return __uint_as_float((u & 0x80000000u) ? (u & 0x7FFFFFFFu) : ~u);
}
__device__ __forceinline__ void edge_ends(const int* __restrict__ ei, int E, int N,
                                          int e, int& s, int& d) {
    if (e < E) { s = ei[e]; d = ei[E + e]; }
    else       { s = e - E; d = e - E; }     // appended self loops
}
__device__ __forceinline__ float eluf(float x) {
    return x > 0.0f ? x : (__expf(x) - 1.0f);
}
__device__ __forceinline__ float leaky(float x) {
    return x > 0.0f ? x : SLOPE * x;
}

// ---------------------------------------------------------------- fill
__global__ void k_fill_u32(unsigned* __restrict__ p, unsigned v, int n) {
    int t = blockIdx.x * blockDim.x + threadIdx.x;
    if (t < n) p[t] = v;
}

// ------------------------------------------------ gather + cvt to f16 rows
__global__ void k_gather_cvt(const float* __restrict__ emb, const int* __restrict__ idx,
                             _Float16* __restrict__ Xh, int N, int D) {
    int t = blockIdx.x * blockDim.x + threadIdx.x;
    if (t >= N * D) return;
    int n = t / D, k = t % D;
    Xh[t] = (_Float16)emb[(size_t)idx[n] * D + k];
}

// ------------------------------------------------ pack W into B-fragment order
// Bf[((tn*KT + kt)*32 + lane)*8 + v] = packed f16 pair (W[kk][col], W[kk+1][col])
// kk = kt*32 + 2v + (v<4?0:8) + 8*half, col = tn*16 + lq   (ISA 7.12.2 B layout)
__global__ void k_pack_w(const float* __restrict__ W, unsigned* __restrict__ Bf,
                         int K, int NOUT) {
    int t = blockIdx.x * blockDim.x + threadIdx.x;
    int KT = K / 32;
    int tot = (NOUT / 16) * KT * 32 * 8;
    if (t >= tot) return;
    int v = t & 7;
    int lane = (t >> 3) & 31;
    int r = t >> 8;
    int kt = r % KT, tn = r / KT;
    int half = lane >> 4, lq = lane & 15;
    int kk = kt * 32 + 2 * v + (v < 4 ? 0 : 8) + 8 * half;
    int col = tn * 16 + lq;
    _Float16 h0 = (_Float16)W[(size_t)kk * NOUT + col];
    _Float16 h1 = (_Float16)W[(size_t)(kk + 1) * NOUT + col];
    Bf[t] = (unsigned)__builtin_bit_cast(unsigned short, h0) |
            ((unsigned)__builtin_bit_cast(unsigned short, h1) << 16);
}

// ---------------------------------------------------------------- WMMA GEMM
// Out[M,NOUT] = Ah(f16,[M,K]) @ W(packed fragments). One wave per 16-row
// M-tile; A fragments live in VGPRs across all N-tiles; K fully unrolled.
template<int K, int NOUT>
__global__ void k_wmma_gemm_t(const _Float16* __restrict__ Ah,
                              const unsigned* __restrict__ Bf,
                              float* __restrict__ Out, int M) {
    constexpr int KT = K / 32;
    int wave = (blockIdx.x * blockDim.x + threadIdx.x) >> 5;
    int lane = threadIdx.x & 31;
    if (wave >= M / 16) return;               // wave-uniform: EXEC all-1s
    int half = lane >> 4, lq = lane & 15;
    int row = wave * 16 + lq;

    const unsigned* arow = (const unsigned*)(Ah + (size_t)row * K);
    v16h afr[KT];
#pragma unroll
    for (int kt = 0; kt < KT; ++kt) {
        v4u lo = *(const v4u*)(arow + kt * 16 + 4 * half);
        v4u hi = *(const v4u*)(arow + kt * 16 + 8 + 4 * half);
        v8u a8;
#pragma unroll
        for (int j = 0; j < 4; ++j) { a8[j] = lo[j]; a8[4 + j] = hi[j]; }
        afr[kt] = __builtin_bit_cast(v16h, a8);
    }

    for (int tn = 0; tn < NOUT / 16; ++tn) {
        const unsigned* bbase = Bf + ((size_t)(tn * KT) * 32 + lane) * 8;
        v8f acc = {};
#pragma unroll
        for (int kt = 0; kt < KT; ++kt) {
            v4u blo = *(const v4u*)(bbase + (size_t)kt * 256);
            v4u bhi = *(const v4u*)(bbase + (size_t)kt * 256 + 4);
            v8u b8;
#pragma unroll
            for (int j = 0; j < 4; ++j) { b8[j] = blo[j]; b8[4 + j] = bhi[j]; }
            v16h b = __builtin_bit_cast(v16h, b8);
            acc = __builtin_amdgcn_wmma_f32_16x16x32_f16(
                false, afr[kt], false, b, (short)0, acc, false, false);
        }
        int ncol = tn * 16 + lq;
#pragma unroll
        for (int r = 0; r < 8; ++r)
            Out[(size_t)(wave * 16 + r + 8 * half) * NOUT + ncol] = acc[r];
    }
}

// ---------------------------------------------------------------- per-node alphas
__global__ void k_alpha(const float* __restrict__ H, const float* __restrict__ att,
                        float* __restrict__ out, int N, int HH, int C) {
    int t = blockIdx.x * blockDim.x + threadIdx.x;
    if (t >= N * HH) return;
    int n = t / HH, h = t % HH;
    const float* hp = H + (size_t)n * HH * C + h * C;
    const float* ap = att + h * C;
    float s = 0.0f;
    for (int c = 0; c < C; ++c) s += hp[c] * ap[c];
    out[t] = s;
}

// ---------------------------------------------------------------- edge passes
__global__ void k_edge_max(const int* __restrict__ ei, int E, int N,
                           const float* __restrict__ AS, const float* __restrict__ AD,
                           unsigned* __restrict__ MX, int HH) {
    int t = blockIdx.x * blockDim.x + threadIdx.x;
    int tot = (E + N) * HH;
    if (t >= tot) return;
    int e = t / HH, h = t % HH;
    int s, d; edge_ends(ei, E, N, e, s, d);
    float l = leaky(AS[s * HH + h] + AD[d * HH + h]);
    atomicMax(&MX[d * HH + h], f2ord(l));
}

__global__ void k_edge_den(const int* __restrict__ ei, int E, int N,
                           const float* __restrict__ AS, const float* __restrict__ AD,
                           const unsigned* __restrict__ MX, float* __restrict__ DEN, int HH) {
    int t = blockIdx.x * blockDim.x + threadIdx.x;
    int tot = (E + N) * HH;
    if (t >= tot) return;
    int e = t / HH, h = t % HH;
    int s, d; edge_ends(ei, E, N, e, s, d);
    float l = leaky(AS[s * HH + h] + AD[d * HH + h]);
    float w = __expf(l - ord2f(MX[d * HH + h]));
    atomicAdd(&DEN[d * HH + h], w);
}

__global__ void k_edge_agg(const int* __restrict__ ei, int E, int N,
                           const float* __restrict__ AS, const float* __restrict__ AD,
                           const unsigned* __restrict__ MX, const float* __restrict__ DEN,
                           const float* __restrict__ H, float* __restrict__ AGG,
                           int HH, int C) {
    int t = blockIdx.x * blockDim.x + threadIdx.x;
    int tot = (E + N) * HH;
    if (t >= tot) return;
    int e = t / HH, h = t % HH;
    int s, d; edge_ends(ei, E, N, e, s, d);
    float l = leaky(AS[s * HH + h] + AD[d * HH + h]);
    float w = __expf(l - ord2f(MX[d * HH + h])) / DEN[d * HH + h];
    const float* hs = H + (size_t)s * HH * C + h * C;
    float* op = AGG + (size_t)d * HH * C + h * C;
    for (int c = 0; c < C; ++c) atomicAdd(&op[c], hs[c] * w);
}

// ---------------------------------------------------------------- bias + ELU
__global__ void k_bias_elu(float* __restrict__ X, const float* __restrict__ b,
                           int N, int HC) {
    int t = blockIdx.x * blockDim.x + threadIdx.x;
    if (t >= N * HC) return;
    X[t] = eluf(X[t] + b[t % HC]);
}

// bias + ELU, output as f16 GEMM operand (layer-2 input)
__global__ void k_bias_elu_cvt(const float* __restrict__ AGG, const float* __restrict__ b,
                               _Float16* __restrict__ Xh, int N, int HC) {
    int t = blockIdx.x * blockDim.x + threadIdx.x;
    if (t >= N * HC) return;
    Xh[t] = (_Float16)eluf(AGG[t] + b[t % HC]);
}

// ---------------------------------------------------------------- gate MLP
__global__ void k_gate(const float* __restrict__ X, const float* __restrict__ gW1,
                       const float* __restrict__ gb1, const float* __restrict__ gW2,
                       const float* __restrict__ gb2, float* __restrict__ GATE, int N) {
    __shared__ float red[64];
    int n = blockIdx.x, j = threadIdx.x;
    float s = gb1[j];
    for (int k = 0; k < 64; ++k) s += X[(size_t)n * 64 + k] * gW1[k * 64 + j];
    s = s > 0.0f ? s : 0.0f;
    red[j] = s * gW2[j];
    __syncthreads();
    if (j == 0) {
        float t = gb2[0];
        for (int k = 0; k < 64; ++k) t += red[k];
        GATE[n] = t;
    }
}

__global__ void k_gmax(const float* __restrict__ GATE, unsigned* __restrict__ GMAX, int N) {
    int t = blockIdx.x * blockDim.x + threadIdx.x;
    if (t < N) atomicMax(GMAX, f2ord(GATE[t]));
}
__global__ void k_gsum(const float* __restrict__ GATE, const unsigned* __restrict__ GMAX,
                       float* __restrict__ GSUM, int N) {
    int t = blockIdx.x * blockDim.x + threadIdx.x;
    if (t < N) atomicAdd(GSUM, __expf(GATE[t] - ord2f(*GMAX)));
}
__global__ void k_gacc(const float* __restrict__ GATE, const unsigned* __restrict__ GMAX,
                       const float* __restrict__ X, float* __restrict__ GACC, int N) {
    int t = blockIdx.x * blockDim.x + threadIdx.x;
    if (t >= N * 64) return;
    int n = t / 64, c = t % 64;
    float w = __expf(GATE[n] - ord2f(*GMAX));
    atomicAdd(&GACC[c], w * X[(size_t)n * 64 + c]);
}
__global__ void k_final(const float* __restrict__ GACC, const float* __restrict__ GSUM,
                        float* __restrict__ out) {
    int c = threadIdx.x;
    if (c < 64) out[c] = GACC[c] / GSUM[0];
}

// ================================================================ launch
extern "C" void kernel_launch(void* const* d_in, const int* in_sizes, int n_in,
                              void* d_out, int out_size, void* d_ws, size_t ws_size,
                              hipStream_t stream) {
    const int*   node_idxs = (const int*)  d_in[0];
    const int*   edge_idx  = (const int*)  d_in[1];
    const float* emb       = (const float*)d_in[2];
    const float* W1        = (const float*)d_in[3];
    const float* att_src1  = (const float*)d_in[4];
    const float* att_dst1  = (const float*)d_in[5];
    const float* b1        = (const float*)d_in[6];
    const float* W2        = (const float*)d_in[7];
    const float* att_src2  = (const float*)d_in[8];
    const float* att_dst2  = (const float*)d_in[9];
    const float* b2        = (const float*)d_in[10];
    const float* gW1       = (const float*)d_in[11];
    const float* gb1       = (const float*)d_in[12];
    const float* gW2       = (const float*)d_in[13];
    const float* gb2       = (const float*)d_in[14];

    const int N = in_sizes[0];          // 20000
    const int E = in_sizes[1] / 2;      // 320000
    const int D = 128, HID = 64, HEADS = 4;
    const int HC1 = HEADS * HID;        // 256
    const int Etot = E + N;

    // ---- workspace layout (float units)
    float* ws = (float*)d_ws;
    size_t o = 0;
    float*    H1   = ws + o; o += (size_t)N * HC1;
    float*    AS1  = ws + o; o += (size_t)N * HEADS;
    float*    AD1  = ws + o; o += (size_t)N * HEADS;
    unsigned* MX1  = (unsigned*)(ws + o); o += (size_t)N * HEADS;
    float*    DEN1 = ws + o; o += (size_t)N * HEADS;
    float*    AGG1 = ws + o; o += (size_t)N * HC1;    // layer-1 aggregation (f32)
    float*    H2   = ws + o; o += (size_t)N * HID;
    float*    AS2  = ws + o; o += (size_t)N;
    float*    AD2  = ws + o; o += (size_t)N;
    unsigned* MX2  = (unsigned*)(ws + o); o += (size_t)N;
    float*    DEN2 = ws + o; o += (size_t)N;
    float*    X3   = ws + o; o += (size_t)N * HID;    // agg2 -> x3 in place
    float*    GATE = ws + o; o += (size_t)N;
    unsigned* GMAX = (unsigned*)(ws + o); o += 1;
    float*    GSUM = ws + o; o += 1;
    float*    GACC = ws + o; o += 64;
    o = (o + 7) & ~(size_t)7;                          // 16B-align f16 operands
    _Float16* Ah1  = (_Float16*)(ws + o); o += (size_t)N * D   / 2;
    _Float16* Ah2  = (_Float16*)(ws + o); o += (size_t)N * HC1 / 2;
    unsigned* Bf1  = (unsigned*)(ws + o); o += (size_t)D   * HC1 / 2;  // 16384 u32
    unsigned* Bf2  = (unsigned*)(ws + o); o += (size_t)HC1 * HID / 2;  //  8192 u32

    const unsigned ORD_NINF = 0x007FFFFFu;   // f2ord(-inf)
    const int BT = 256;

    // ---- init accumulators (graph-capture-safe)
    hipMemsetAsync(DEN1, 0, (size_t)(N * HEADS + N * HC1) * 4, stream); // DEN1 + AGG1
    hipMemsetAsync(DEN2, 0, (size_t)(N + N * HID) * 4, stream);         // DEN2 + X3
    hipMemsetAsync(GSUM, 0, (size_t)65 * 4, stream);                    // GSUM + GACC
    k_fill_u32<<<CEILDIV(N * HEADS, BT), BT, 0, stream>>>(MX1, ORD_NINF, N * HEADS);
    k_fill_u32<<<CEILDIV(N, BT), BT, 0, stream>>>(MX2, ORD_NINF, N);
    k_fill_u32<<<1, 32, 0, stream>>>(GMAX, ORD_NINF, 1);

    // ---- operand staging
    k_gather_cvt<<<CEILDIV(N * D, BT), BT, 0, stream>>>(emb, node_idxs, Ah1, N, D);
    k_pack_w<<<CEILDIV((HC1 / 16) * (D / 32) * 256, BT), BT, 0, stream>>>(W1, Bf1, D, HC1);
    k_pack_w<<<CEILDIV((HID / 16) * (HC1 / 32) * 256, BT), BT, 0, stream>>>(W2, Bf2, HC1, HID);

    // ---- layer 1: H1 = x @ W1  (WMMA)
    k_wmma_gemm_t<128, 256><<<CEILDIV(N / 16, 8), BT, 0, stream>>>(Ah1, Bf1, H1, N);
    k_alpha<<<CEILDIV(N * HEADS, BT), BT, 0, stream>>>(H1, att_src1, AS1, N, HEADS, HID);
    k_alpha<<<CEILDIV(N * HEADS, BT), BT, 0, stream>>>(H1, att_dst1, AD1, N, HEADS, HID);
    k_edge_max<<<CEILDIV(Etot * HEADS, BT), BT, 0, stream>>>(edge_idx, E, N, AS1, AD1, MX1, HEADS);
    k_edge_den<<<CEILDIV(Etot * HEADS, BT), BT, 0, stream>>>(edge_idx, E, N, AS1, AD1, MX1, DEN1, HEADS);
    k_edge_agg<<<CEILDIV(Etot * HEADS, BT), BT, 0, stream>>>(edge_idx, E, N, AS1, AD1, MX1, DEN1,
                                                             H1, AGG1, HEADS, HID);
    k_bias_elu_cvt<<<CEILDIV(N * HC1, BT), BT, 0, stream>>>(AGG1, b1, Ah2, N, HC1);

    // ---- layer 2: H2 = x2 @ W2 (WMMA)
    k_wmma_gemm_t<256, 64><<<CEILDIV(N / 16, 8), BT, 0, stream>>>(Ah2, Bf2, H2, N);
    k_alpha<<<CEILDIV(N, BT), BT, 0, stream>>>(H2, att_src2, AS2, N, 1, HID);
    k_alpha<<<CEILDIV(N, BT), BT, 0, stream>>>(H2, att_dst2, AD2, N, 1, HID);
    k_edge_max<<<CEILDIV(Etot, BT), BT, 0, stream>>>(edge_idx, E, N, AS2, AD2, MX2, 1);
    k_edge_den<<<CEILDIV(Etot, BT), BT, 0, stream>>>(edge_idx, E, N, AS2, AD2, MX2, DEN2, 1);
    k_edge_agg<<<CEILDIV(Etot, BT), BT, 0, stream>>>(edge_idx, E, N, AS2, AD2, MX2, DEN2,
                                                     H2, X3, 1, HID);
    k_bias_elu<<<CEILDIV(N * HID, BT), BT, 0, stream>>>(X3, b2, N, HID);

    // ---- attentional pooling
    k_gate<<<N, 64, 0, stream>>>(X3, gW1, gb1, gW2, gb2, GATE, N);
    k_gmax<<<CEILDIV(N, BT), BT, 0, stream>>>(GATE, GMAX, N);
    k_gsum<<<CEILDIV(N, BT), BT, 0, stream>>>(GATE, GMAX, GSUM, N);
    k_gacc<<<CEILDIV(N * 64, BT), BT, 0, stream>>>(GATE, GMAX, X3, GACC, N);
    k_final<<<1, 64, 0, stream>>>(GACC, GSUM, (float*)d_out);
}
